// MultiHeadAttention_7894149890102
// MI455X (gfx1250) — matrix operand order repaired and checked by reading, not compiled, not run on hardware
//
#include <hip/hip_runtime.h>
#include <hip/hip_bf16.h>

typedef unsigned short u16;
typedef __attribute__((ext_vector_type(16))) __bf16 v16bf;
typedef __attribute__((ext_vector_type(8)))  float  v8f;

union Frag16 { v16bf v; u16 u[16]; uint4 q[2]; };

static __device__ inline u16 f2bf(float f) {
  unsigned int u = __float_as_uint(f);
  unsigned int r = (u + 0x7FFFu + ((u >> 16) & 1u)) >> 16;
  return (u16)r;
}

static __device__ inline v8f wmma_bf16(const Frag16& a, const Frag16& b, v8f c) {
  return __builtin_amdgcn_wmma_f32_16x16x32_bf16(false, a.v, false, b.v,
                                                 (short)0, c, false, false);
}

static __device__ inline v8f zero8() {
  v8f z;
#pragma unroll
  for (int i = 0; i < 8; ++i) z[i] = 0.0f;
  return z;
}

// Async DMA of 32 contiguous bytes (2 x B128) global -> LDS, tracked by ASYNCcnt.
// Per CDNA5 ISA, INST_OFFSET is added to both the LDS and the global address,
// so one base pair + offset:16 works.
static __device__ inline void async_copy32(unsigned lds_off, const u16* g) {
  asm volatile("global_load_async_to_lds_b128 %0, %1, off\n\t"
               "global_load_async_to_lds_b128 %0, %1, off offset:16"
               :: "v"(lds_off), "v"(g) : "memory");
}
static __device__ inline void wait_async0() {
  asm volatile("s_wait_asynccnt 0x0" ::: "memory");
}
static __device__ inline unsigned lds_addr(const void* p) {
  return (unsigned)(unsigned long long)p;   // flat->LDS uses addr[31:0]
}

// ---------------------------------------------------------------- pack kernels

__global__ void cvt_bf16_kernel(const float* __restrict__ in, u16* __restrict__ out, int n) {
  for (int i = blockIdx.x * blockDim.x + threadIdx.x; i < n; i += gridDim.x * blockDim.x)
    out[i] = f2bf(in[i]);
}

// W: [H=16][K=1024][D=64] -> WT[n=h*64+d][k=0..1023]  (n-major, k-contiguous)
__global__ void pack_wT_kernel(const float* __restrict__ W, u16* __restrict__ WT) {
  for (int i = blockIdx.x * blockDim.x + threadIdx.x; i < (1 << 20); i += gridDim.x * blockDim.x) {
    int n = i >> 10, k = i & 1023;
    int h = n >> 6, d = n & 63;
    WT[i] = f2bf(W[(h << 16) + (k << 6) + d]);
  }
}

// rp_k_table [127][64] -> [128][64] bf16 (row 127 zero). Serves as Bt for rq GEMM.
__global__ void pack_rpk_kernel(const float* __restrict__ T, u16* __restrict__ out) {
  for (int i = blockIdx.x * blockDim.x + threadIdx.x; i < 128 * 64; i += gridDim.x * blockDim.x) {
    int t = i >> 6, d = i & 63;
    out[i] = (t < 127) ? f2bf(T[t * 64 + d]) : (u16)0;
  }
}

// rp_v_table [127][64] -> transposed [64][128] bf16 (col 127 zero).
__global__ void pack_rpv_kernel(const float* __restrict__ T, u16* __restrict__ out) {
  for (int i = blockIdx.x * blockDim.x + threadIdx.x; i < 64 * 128; i += gridDim.x * blockDim.x) {
    int d = i >> 7, t = i & 127;
    out[i] = (t < 127) ? f2bf(T[t * 64 + d]) : (u16)0;
  }
}

// ---------------------------------------------------------------- generic WMMA GEMM
// C[M,N] = A[M,K](bf16) @ Bt[N,K](bf16, n-major) + bias[N]
// mode 0: f32 row-major; mode 1: bf16 head-split [B,H,S,64]; mode 2: bf16 [B,H,64,S]
// Double-buffered LDS tiles filled by async global->LDS DMA (ASYNCcnt).
__global__ __launch_bounds__(256)
void gemm_bf16_kernel(const u16* __restrict__ A, const u16* __restrict__ Bt,
                      const float* __restrict__ bias, void* __restrict__ Cout,
                      int M, int N, int K, int mode) {
  __shared__ u16 ldsA[2][128 * 32];
  __shared__ u16 ldsB[2][128 * 32];
  const int tid  = threadIdx.x;
  const int lane = tid & 31;
  const int wave = tid >> 5;
  const int wm = wave >> 1, wn = wave & 1;      // 4 x 2 wave grid
  const int m0 = blockIdx.y * 128;
  const int n0 = blockIdx.x * 128;
  const int mW = wm * 32, nW = wn * 64;
  const int l15 = lane & 15, half = lane >> 4;
  const int sRow = tid >> 1;
  const int sCol = (tid & 1) * 16;

  v8f acc[2][4];
#pragma unroll
  for (int mi = 0; mi < 2; ++mi)
#pragma unroll
    for (int ni = 0; ni < 4; ++ni) acc[mi][ni] = zero8();

  const int nSteps = K >> 5;
  const u16* gA0 = A + (size_t)(m0 + sRow) * K + sCol;
  const u16* gB0 = Bt + (size_t)(n0 + sRow) * K + sCol;

  // prologue: stage step 0 into buffer 0
  async_copy32(lds_addr(&ldsA[0][sRow * 32 + sCol]), gA0);
  async_copy32(lds_addr(&ldsB[0][sRow * 32 + sCol]), gB0);

  for (int t = 0; t < nSteps; ++t) {
    const int buf = t & 1;
    wait_async0();          // my async fills for buf are done
    __syncthreads();        // everyone's fills done; nobody still reads buf^1
    if (t + 1 < nSteps) {
      async_copy32(lds_addr(&ldsA[buf ^ 1][sRow * 32 + sCol]), gA0 + (t + 1) * 32);
      async_copy32(lds_addr(&ldsB[buf ^ 1][sRow * 32 + sCol]), gB0 + (t + 1) * 32);
    }
    Frag16 a[2], b[4];
#pragma unroll
    for (int mi = 0; mi < 2; ++mi) {
      const u16* r = &ldsA[buf][(mW + mi * 16 + l15) * 32];
      a[mi].q[0] = *(const uint4*)(r + half * 8);
      a[mi].q[1] = *(const uint4*)(r + 16 + half * 8);
    }
#pragma unroll
    for (int ni = 0; ni < 4; ++ni) {
      const u16* r = &ldsB[buf][(nW + ni * 16 + l15) * 32];
      b[ni].q[0] = *(const uint4*)(r + half * 8);
      b[ni].q[1] = *(const uint4*)(r + 16 + half * 8);
    }
#pragma unroll
    for (int mi = 0; mi < 2; ++mi)
#pragma unroll
      for (int ni = 0; ni < 4; ++ni)
        acc[mi][ni] = wmma_bf16(a[mi], b[ni], acc[mi][ni]);
  }

#pragma unroll
  for (int mi = 0; mi < 2; ++mi)
#pragma unroll
    for (int ni = 0; ni < 4; ++ni) {
      int n = n0 + nW + ni * 16 + l15;
      float bv = bias ? bias[n] : 0.0f;
#pragma unroll
      for (int r = 0; r < 8; ++r) {
        int m = m0 + mW + mi * 16 + half * 8 + r;
        float val = acc[mi][ni][r] + bv;
        if (mode == 0) {
          ((float*)Cout)[(size_t)m * N + n] = val;
        } else if (mode == 1) {
          int bb = m >> 10, s = m & 1023, h = n >> 6, d = n & 63;
          ((u16*)Cout)[(((size_t)(bb * 16 + h) << 10) + s) * 64 + d] = f2bf(val);
        } else {
          int bb = m >> 10, s = m & 1023, h = n >> 6, d = n & 63;
          ((u16*)Cout)[((size_t)(bb * 16 + h) * 64 + d) * 1024 + s] = f2bf(val);
        }
      }
    }
}

// ---------------------------------------------------------------- attention
// One workgroup = one (b,h), 64 query rows. 8 waves (256 threads):
//  wave = (kvHalf<<2) | rowGroup: rowGroup picks 16 query rows, kvHalf splits
//  the key range / softmax columns / output n-tiles.
// Dynamic LDS: sc[64][1024] f32 (256KB) + wsum[64][128] f32 (32KB) = 288KB.
__global__ __launch_bounds__(256)
void attn_kernel(const u16* __restrict__ q_ws, const u16* __restrict__ k_ws,
                 const u16* __restrict__ vT_ws, const float* __restrict__ rq_ws,
                 const u16* __restrict__ rp_vT, u16* __restrict__ hidden_bf) {
  extern __shared__ float smem[];
  float* sc   = smem;               // [64][1024]
  float* wsum = smem + 64 * 1024;   // [64][128], col 127 stays 0
  __shared__ float pmax[64 * 2];
  __shared__ float psum[64 * 2];

  const int tid = threadIdx.x, lane = tid & 31, wave = tid >> 5;
  const int rg  = wave & 3;                  // row group: rows rg*16..+15
  const int kvH = wave >> 2;                 // 0/1: key-range half
  const int bh = blockIdx.x >> 4;            // b*16+h
  const int qBase = (blockIdx.x & 15) * 64;
  const int bb = bh >> 4, h = bh & 15;
  const int l15 = lane & 15, half = lane >> 4;

  for (int i = tid; i < 64 * 128; i += 256) wsum[i] = 0.0f;

  // ---- phase 1: scores = (Q K^T + rq-gather) * 1/8   (waves split 1024 keys)
  {
    Frag16 A0, A1;
    const u16* qrow = q_ws + (size_t)(bh * 1024 + qBase + rg * 16 + l15) * 64;
    A0.q[0] = *(const uint4*)(qrow + half * 8);
    A0.q[1] = *(const uint4*)(qrow + 16 + half * 8);
    A1.q[0] = *(const uint4*)(qrow + 32 + half * 8);
    A1.q[1] = *(const uint4*)(qrow + 48 + half * 8);
    for (int t = 0; t < 32; ++t) {
      int kv = (kvH * 32 + t) * 16;
      const u16* krow = k_ws + (size_t)(bh * 1024 + kv + l15) * 64;
      Frag16 B0, B1;
      B0.q[0] = *(const uint4*)(krow + half * 8);
      B0.q[1] = *(const uint4*)(krow + 16 + half * 8);
      B1.q[0] = *(const uint4*)(krow + 32 + half * 8);
      B1.q[1] = *(const uint4*)(krow + 48 + half * 8);
      v8f c = zero8();
      c = wmma_bf16(A0, B0, c);
      c = wmma_bf16(A1, B1, c);
      int j = kv + l15;
#pragma unroll
      for (int r = 0; r < 8; ++r) {
        int iloc = rg * 16 + half * 8 + r;
        int ig = qBase + iloc;
        int dd = j - ig;
        dd = dd < -63 ? -63 : (dd > 63 ? 63 : dd);
        float bias = rq_ws[(size_t)(bh * 1024 + ig) * 128 + dd + 63];
        sc[iloc * 1024 + j] = (c[r] + bias) * 0.125f;
      }
    }
  }
  __syncthreads();

  // ---- phase 2: row softmax (unnormalized) + wsum scatter. 4 lanes per row.
  {
    const int row = rg * 16 + l15;
    const int ig = qBase + row;
    const int q4 = kvH * 2 + half;           // column quarter 0..3
    float* r = sc + row * 1024 + q4 * 256;
    float mx = -3.4e38f;
    for (int j = 0; j < 256; j += 4) {
      float4 v4 = *(const float4*)(r + j);
      mx = fmaxf(mx, fmaxf(fmaxf(v4.x, v4.y), fmaxf(v4.z, v4.w)));
    }
    mx = fmaxf(mx, __shfl_xor(mx, 16));      // combine this wave's 2 quarters
    if (half == 0) pmax[row * 2 + kvH] = mx;
    __syncthreads();
    const float fm = fmaxf(pmax[row * 2], pmax[row * 2 + 1]);
    float sum = 0.0f;
    const int jg0 = q4 * 256;
    for (int j = 0; j < 256; ++j) {
      float e = __expf(r[j] - fm);
      r[j] = e;
      sum += e;
      int dd = (jg0 + j) - ig;
      dd = dd < -63 ? -63 : (dd > 63 ? 63 : dd);
      atomicAdd(&wsum[row * 128 + dd + 63], e);   // ds_add_f32
    }
    sum += __shfl_xor(sum, 16);
    if (half == 0) psum[row * 2 + kvH] = sum;
  }
  __syncthreads();

  // ---- phase 3: out = (w @ V + wsum @ rp_v) / l   (waves split 4 n-tiles 2+2)
  {
    v8f acc[2];
    acc[0] = zero8(); acc[1] = zero8();
    const int rowA = rg * 16 + l15;

    for (int ks = 0; ks < 32; ++ks) {
      Frag16 a;
      const float* srow = sc + rowA * 1024 + ks * 32;
#pragma unroll
      for (int e = 0; e < 8; ++e) a.u[e] = f2bf(srow[half * 8 + e]);
#pragma unroll
      for (int e = 0; e < 8; ++e) a.u[8 + e] = f2bf(srow[16 + half * 8 + e]);
#pragma unroll
      for (int ni = 0; ni < 2; ++ni) {
        int d = (kvH * 2 + ni) * 16 + l15;
        const u16* vrow = vT_ws + (size_t)(bh * 64 + d) * 1024 + ks * 32 + half * 8;
        Frag16 b;
        b.q[0] = *(const uint4*)vrow;
        b.q[1] = *(const uint4*)(vrow + 16);
        acc[ni] = wmma_bf16(a, b, acc[ni]);
      }
    }
    for (int ks = 0; ks < 4; ++ks) {
      Frag16 a;
      const float* srow = wsum + rowA * 128 + ks * 32;
#pragma unroll
      for (int e = 0; e < 8; ++e) a.u[e] = f2bf(srow[half * 8 + e]);
#pragma unroll
      for (int e = 0; e < 8; ++e) a.u[8 + e] = f2bf(srow[16 + half * 8 + e]);
#pragma unroll
      for (int ni = 0; ni < 2; ++ni) {
        int d = (kvH * 2 + ni) * 16 + l15;
        const u16* vrow = rp_vT + d * 128 + ks * 32 + half * 8;
        Frag16 b;
        b.q[0] = *(const uint4*)vrow;
        b.q[1] = *(const uint4*)(vrow + 16);
        acc[ni] = wmma_bf16(a, b, acc[ni]);
      }
    }
#pragma unroll
    for (int ni = 0; ni < 2; ++ni) {
      int d = (kvH * 2 + ni) * 16 + l15;
#pragma unroll
      for (int r = 0; r < 8; ++r) {
        int iloc = rg * 16 + half * 8 + r;
        float l = psum[iloc * 2] + psum[iloc * 2 + 1];
        float val = acc[ni][r] / l;
        int s = qBase + iloc;
        hidden_bf[(size_t)(bb * 1024 + s) * 1024 + h * 64 + d] = f2bf(val);
      }
    }
  }
}

// ---------------------------------------------------------------- host

extern "C" void kernel_launch(void* const* d_in, const int* in_sizes, int n_in,
                              void* d_out, int out_size, void* d_ws, size_t ws_size,
                              hipStream_t stream) {
  (void)in_sizes; (void)n_in; (void)out_size;
  const float* query = (const float*)d_in[0];
  const float* key   = (const float*)d_in[1];
  const float* value = (const float*)d_in[2];
  const float* Wq    = (const float*)d_in[3];
  const float* Wk    = (const float*)d_in[4];
  const float* Wv    = (const float*)d_in[5];
  const float* bq    = (const float*)d_in[6];
  const float* bk    = (const float*)d_in[7];
  const float* bv    = (const float*)d_in[8];
  const float* rpk   = (const float*)d_in[9];
  const float* rpv   = (const float*)d_in[10];
  const float* Wfc   = (const float*)d_in[11];
  const float* bfc   = (const float*)d_in[12];
  float* out = (float*)d_out;

  char* ws = (char*)d_ws;
  size_t cur = 0;
  auto alloc = [&](size_t bytes) -> char* {
    char* p = ws + cur;
    cur += (bytes + 255) & ~(size_t)255;
    return p;
  };
  const size_t ACT = (size_t)4096 * 1024;   // activation elements
  u16* q_act  = (u16*)alloc(ACT * 2);
  u16* k_act  = (u16*)alloc(ACT * 2);
  u16* v_act  = (u16*)alloc(ACT * 2);
  u16* WqT    = (u16*)alloc((size_t)1024 * 1024 * 2);
  u16* WkT    = (u16*)alloc((size_t)1024 * 1024 * 2);
  u16* WvT    = (u16*)alloc((size_t)1024 * 1024 * 2);
  u16* Wfc_bf = (u16*)alloc((size_t)1024 * 1024 * 2);
  u16* rpk_bf = (u16*)alloc((size_t)128 * 64 * 2);
  u16* rpvT   = (u16*)alloc((size_t)64 * 128 * 2);
  u16* q_ws   = (u16*)alloc(ACT * 2);       // [B,H,S,64] bf16
  u16* k_ws   = (u16*)alloc(ACT * 2);
  u16* vT_ws  = (u16*)alloc(ACT * 2);       // [B,H,64,S] bf16
  float* rq_ws = (float*)alloc((size_t)65536 * 128 * 4);
  u16* hid_bf = (u16*)alloc(ACT * 2);
  (void)ws_size;  // ~96 MB required

  // pack
  cvt_bf16_kernel<<<4096, 256, 0, stream>>>(query, q_act, (int)ACT);
  cvt_bf16_kernel<<<4096, 256, 0, stream>>>(key,   k_act, (int)ACT);
  cvt_bf16_kernel<<<4096, 256, 0, stream>>>(value, v_act, (int)ACT);
  cvt_bf16_kernel<<<1024, 256, 0, stream>>>(Wfc, Wfc_bf, 1 << 20);
  pack_wT_kernel<<<1024, 256, 0, stream>>>(Wq, WqT);
  pack_wT_kernel<<<1024, 256, 0, stream>>>(Wk, WkT);
  pack_wT_kernel<<<1024, 256, 0, stream>>>(Wv, WvT);
  pack_rpk_kernel<<<32, 256, 0, stream>>>(rpk, rpk_bf);
  pack_rpv_kernel<<<32, 256, 0, stream>>>(rpv, rpvT);

  // projections
  dim3 gProj(1024 / 128, 4096 / 128);
  gemm_bf16_kernel<<<gProj, 256, 0, stream>>>(q_act, WqT, bq, q_ws,  4096, 1024, 1024, 1);
  gemm_bf16_kernel<<<gProj, 256, 0, stream>>>(k_act, WkT, bk, k_ws,  4096, 1024, 1024, 1);
  gemm_bf16_kernel<<<gProj, 256, 0, stream>>>(v_act, WvT, bv, vT_ws, 4096, 1024, 1024, 2);

  // rq[b,h,s,t] = q . rp_k_table[t]  (N padded to 128)
  dim3 gRq(1, 65536 / 128);
  gemm_bf16_kernel<<<gRq, 256, 0, stream>>>(q_ws, rpk_bf, nullptr, rq_ws, 65536, 128, 64, 0);

  // attention (288 KB dynamic LDS per workgroup, 8 waves)
  attn_kernel<<<64 * 16, 256, 294912, stream>>>(q_ws, k_ws, vT_ws, rq_ws, rpvT, hid_bf);

  // final FC: hidden @ Wfc^T + bfc (Wfc is already n-major,k-contiguous)
  gemm_bf16_kernel<<<gProj, 256, 0, stream>>>(hid_bf, Wfc_bf, bfc, out, 4096, 1024, 1024, 0);
}
